// LightweightDeformAlign_1726576853504
// MI455X (gfx1250) — compile-verified, matrix operand order checked by reading, not run on hardware
//
#include <hip/hip_runtime.h>
#include <stdint.h>

typedef __attribute__((ext_vector_type(16))) _Float16 v16h;
typedef __attribute__((ext_vector_type(8)))  float    v8f;

constexpr int BATCH = 8, HH = 80, WW = 80, HWs = HH * WW;
constexpr int CCH = 128, MID = 64, OFFC = 72, NG = 4, CG = 32;
constexpr int MTOT = BATCH * HWs;   // 51200 output pixels
constexpr int LDK = 40;             // 32 k + 8 pad halves -> 80B rows, 16B-aligned stores

// ---- WMMA 16-bit A/B fragment load from an LDS row (k-contiguous) ----
__device__ __forceinline__ v16h frag_ld(const _Float16* rowbase) {
  const int lane = threadIdx.x & 31;
  const int kb = (lane >> 4) << 3;
  union { v16h v; uint32_t u[8]; } r;
#pragma unroll
  for (int i = 0; i < 8; ++i) {
    const int k = kb + ((i >> 2) << 4) + ((i & 3) << 1);
    r.u[i] = *(const uint32_t*)(rowbase + k);
  }
  return r.v;
}

__device__ __forceinline__ void pack8_store(_Float16* dst, const float* s) {
  union { _Float16 h[8]; uint4 q; } p;
#pragma unroll
  for (int i = 0; i < 8; ++i) p.h[i] = (_Float16)s[i];
  *(uint4*)dst = p.q;
}

__device__ __forceinline__ v8f wmma16(v16h a, v16h b, v8f c) {
  return __builtin_amdgcn_wmma_f32_16x16x32_f16(false, a, false, b, (short)0, c, false, false);
}

// Async contiguous 16B global->LDS copy (ASYNCcnt-tracked, CDNA5 ISA)
__device__ __forceinline__ void async_copy_b128(uint32_t lds_off, const _Float16* gsrc) {
  asm volatile("global_load_async_to_lds_b128 %0, %1, off"
               :: "v"(lds_off), "v"((uint64_t)(uintptr_t)gsrc) : "memory");
}
__device__ __forceinline__ void async_wait0() {
  asm volatile("s_wait_asynccnt 0x0" ::: "memory");
}

// ============ one-time weight pre-pack: f32 -> f16 in per-chunk B-tile order ============
// w1p: [t][cbi(8)][nl(64)][cl(32)]           (conv1: Cin=256, Cout=64, 1 N-tile)
// w2p: [t][cbi(2)][ntile(2)][nl(64)][cl(32)] (conv2: Cin=64, Cout=72 padded to 128)
// dwp: [g][t][o(32)][c(32)]                  (deform group weights)
__global__ __launch_bounds__(256)
void pack_weights(const float* __restrict__ w1, const float* __restrict__ w2,
                  const float* __restrict__ dw,
                  _Float16* __restrict__ w1p, _Float16* __restrict__ w2p,
                  _Float16* __restrict__ dwp) {
  const int idx = blockIdx.x * 256 + threadIdx.x;
  constexpr int W1N = 9 * 8 * 64 * 32;
  constexpr int W2N = 9 * 2 * 2 * 64 * 32;
  constexpr int DWN = 4 * 9 * 32 * 32;
  if (idx < W1N) {
    const int cl = idx & 31, nl = (idx >> 5) & 63, cbi = (idx >> 11) & 7, t = idx >> 14;
    w1p[idx] = (_Float16)w1[(size_t)nl * (256 * 9) + (cbi * 32 + cl) * 9 + t];
  }
  if (idx < W2N) {
    const int cl = idx & 31, nl = (idx >> 5) & 63, ntile = (idx >> 11) & 1,
              cbi = (idx >> 12) & 1, t = idx >> 13;
    const int n = ntile * 64 + nl;
    w2p[idx] = (n < OFFC) ? (_Float16)w2[(size_t)n * (64 * 9) + (cbi * 32 + cl) * 9 + t]
                          : (_Float16)0.0f;
  }
  if (idx < DWN) {
    const int c = idx & 31, o = (idx >> 5) & 31, rest = idx >> 10;
    const int t = rest % 9, g = rest / 9;
    dwp[idx] = (_Float16)dw[((size_t)(g * CG + o) * CG + c) * 9 + t];
  }
}

// ===================== implicit-GEMM conv3x3 (pad 1, stride 1) =====================
// K order: (tap, c-block). Pad handled by clamped address + {0,1} mask multiply
// (no divergence). A pointer advances incrementally; c-loop split per source tensor.
__global__ __launch_bounds__(256, 2)
void conv3x3_wmma(const float* __restrict__ in0, const float* __restrict__ in1, int Csplit,
                  const _Float16* __restrict__ wpk, const float* __restrict__ bias,
                  const float* __restrict__ scale_ptr, float* __restrict__ out,
                  int Cin, int Cout, int ntiles, int relu) {
  __shared__ _Float16 Alds[128 * LDK];
  __shared__ _Float16 Blds[64 * LDK];

  const int tid  = threadIdx.x;
  const int lane = tid & 31;
  const int wv   = tid >> 5;
  const int mt   = wv & 3;            // 4 M-waves x 32 rows
  const int nt   = wv >> 2;           // 2 N-waves x 32 cols
  const int M0   = blockIdx.x * 128;
  const int N0   = blockIdx.y * 64;
  const int nCb  = Cin >> 5;
  const int nCb0 = Csplit >> 5;       // chunks sourced from in0
  const float scale = scale_ptr ? scale_ptr[0] : 1.0f;

  const int arow = tid & 127;
  const int ac0  = (tid >> 7) * 16;
  const int am   = M0 + arow;
  const int ab   = am / HWs;
  const int arem = am - ab * HWs;
  const int ay   = arem / WW;
  const int ax   = arem - ay * WW;

  const uint32_t bldsoff =
      (uint32_t)(uintptr_t)&Blds[(tid >> 2) * LDK + (tid & 3) * 8];
  _Float16* const a_dst = &Alds[arow * LDK + ac0];
  const _Float16* const a_frag = &Alds[(mt * 32 + (lane & 15)) * LDK];
  const _Float16* const b_frag = &Blds[(nt * 32 + (lane & 15)) * LDK];

  v8f acc00 = {}, acc01 = {}, acc10 = {}, acc11 = {};

#pragma unroll
  for (int t = 0; t < 9; ++t) {
    const int ky = t / 3 - 1, kx = t % 3 - 1;
    const int yy = ay + ky, xx = ax + kx;
    const bool valid = (yy >= 0) & (yy < HH) & (xx >= 0) & (xx < WW);
    const float maskf = valid ? 1.0f : 0.0f;
    const int soff = min(max(yy, 0), HH - 1) * WW + min(max(xx, 0), WW - 1);

    // chunk body: async B pull + masked A stage + 4 WMMAs
    auto do_chunk = [&](const float* p, const _Float16* bsrc) {
      async_copy_b128(bldsoff, bsrc);
      float v[16];
#pragma unroll
      for (int i = 0; i < 16; ++i) v[i] = p[(size_t)i * HWs];
#pragma unroll
      for (int i = 0; i < 16; ++i) v[i] *= maskf;
      pack8_store(a_dst, v);
      pack8_store(a_dst + 8, v + 8);
      async_wait0();
      __syncthreads();
      const v16h a0 = frag_ld(a_frag);
      const v16h a1 = frag_ld(a_frag + 16 * LDK);
      const v16h b0 = frag_ld(b_frag);
      const v16h b1 = frag_ld(b_frag + 16 * LDK);
      acc00 = wmma16(a0, b0, acc00);
      acc01 = wmma16(a0, b1, acc01);
      acc10 = wmma16(a1, b0, acc10);
      acc11 = wmma16(a1, b1, acc11);
      __syncthreads();
    };

    const size_t bstep = (size_t)ntiles << 11;
    const _Float16* bsrc =
        wpk + ((size_t)(t * nCb * ntiles + blockIdx.y) << 11) + tid * 8;

    const float* p0 = in0 + ((size_t)ab * Csplit + ac0) * HWs + soff;
    for (int cbi = 0; cbi < nCb0; ++cbi) {
      do_chunk(p0, bsrc);
      p0 += (size_t)32 * HWs;
      bsrc += bstep;
    }
    if (nCb > nCb0) {
      const float* p1 = in1 + ((size_t)ab * (Cin - Csplit) + ac0) * HWs + soff;
      for (int cbi = nCb0; cbi < nCb; ++cbi) {
        do_chunk(p1, bsrc);
        p1 += (size_t)32 * HWs;
        bsrc += bstep;
      }
    }
  }

  const int nloc = lane & 15;
  const int mhi  = (lane >> 4) << 3;
  const v8f accs[2][2] = {{acc00, acc01}, {acc10, acc11}};
#pragma unroll
  for (int i = 0; i < 2; ++i) {
#pragma unroll
    for (int j = 0; j < 2; ++j) {
      const int n = N0 + nt * 32 + j * 16 + nloc;
      if (n < Cout) {
        const float bv = bias[n];
#pragma unroll
        for (int r = 0; r < 8; ++r) {
          const int m = M0 + mt * 32 + i * 16 + mhi + r;
          const int b = m / HWs;
          const int rem = m - b * HWs;
          float vv = accs[i][j][r] + bv;
          if (relu) vv = fmaxf(vv, 0.0f);
          out[((size_t)b * Cout + n) * HWs + rem] = vv * scale;
        }
      }
    }
  }
}

// ===================== deformable conv3x3 (groups=4) =====================
__global__ __launch_bounds__(256, 2)
void deform_wmma(const float* __restrict__ x, const float* __restrict__ offs,
                 const _Float16* __restrict__ dwp, const float* __restrict__ db,
                 float* __restrict__ out) {
  __shared__ _Float16 Alds[256 * LDK];
  __shared__ _Float16 Blds[32 * LDK];

  const int tid  = threadIdx.x;
  const int lane = tid & 31;
  const int wv   = tid >> 5;
  const int M0   = blockIdx.x * 256;
  const int g    = blockIdx.y;

  const int am   = M0 + tid;
  const int ab   = am / HWs;
  const int arem = am - ab * HWs;
  const int ay   = arem / WW;
  const int ax   = arem - ay * WW;
  const float* xg = x + ((size_t)ab * CCH + g * CG) * HWs;

  const uint32_t bldsoff =
      (uint32_t)(uintptr_t)&Blds[(tid >> 2) * LDK + (tid & 3) * 8];

  v8f acc00 = {}, acc01 = {}, acc10 = {}, acc11 = {};

#pragma unroll
  for (int t = 0; t < 9; ++t) {
    if (tid < 128) {
      const _Float16* bsrc = dwp + ((size_t)(g * 9 + t) << 10) + tid * 8;
      async_copy_b128(bldsoff, bsrc);
    }

    const int ky = t / 3 - 1, kx = t % 3 - 1;
    const size_t ob = ((size_t)ab * OFFC + g * 18 + 2 * t) * HWs + ay * WW + ax;
    const float oy = offs[ob];
    const float ox = offs[ob + HWs];
    const float sy = (float)(ay + ky) + oy;
    const float sx = (float)(ax + kx) + ox;
    const float fy = floorf(sy), fx = floorf(sx);
    const int y0 = (int)fy, x0 = (int)fx;
    const float wy = sy - fy, wx = sx - fx;
    const bool vy0 = (y0 >= 0) & (y0 < HH), vy1 = (y0 + 1 >= 0) & (y0 + 1 < HH);
    const bool vx0 = (x0 >= 0) & (x0 < WW), vx1 = (x0 + 1 >= 0) & (x0 + 1 < WW);
    const float m00 = (vy0 & vx0) ? (1.f - wy) * (1.f - wx) : 0.f;
    const float m01 = (vy0 & vx1) ? (1.f - wy) * wx         : 0.f;
    const float m10 = (vy1 & vx0) ? wy * (1.f - wx)         : 0.f;
    const float m11 = (vy1 & vx1) ? wy * wx                 : 0.f;
    const int y0c = min(max(y0, 0), HH - 1), y1c = min(max(y0 + 1, 0), HH - 1);
    const int x0c = min(max(x0, 0), WW - 1), x1c = min(max(x0 + 1, 0), WW - 1);
    const int i00 = y0c * WW + x0c, i01 = y0c * WW + x1c;
    const int i10 = y1c * WW + x0c, i11 = y1c * WW + x1c;

#pragma unroll
    for (int cblk = 0; cblk < 4; ++cblk) {
      float c00[8], c01[8], c10[8], c11[8], s[8];
#pragma unroll
      for (int i = 0; i < 8; ++i) {
        const float* ch = xg + (size_t)(cblk * 8 + i) * HWs;
        c00[i] = ch[i00]; c01[i] = ch[i01]; c10[i] = ch[i10]; c11[i] = ch[i11];
      }
#pragma unroll
      for (int i = 0; i < 8; ++i)
        s[i] = c00[i] * m00 + c01[i] * m01 + c10[i] * m10 + c11[i] * m11;
      pack8_store(&Alds[tid * LDK + cblk * 8], s);
    }
    async_wait0();
    __syncthreads();

    const _Float16* abase = &Alds[(wv * 32 + (lane & 15)) * LDK];
    const v16h a0 = frag_ld(abase);
    const v16h a1 = frag_ld(abase + 16 * LDK);
    const _Float16* bbase = &Blds[(lane & 15) * LDK];
    const v16h b0 = frag_ld(bbase);
    const v16h b1 = frag_ld(bbase + 16 * LDK);
    acc00 = wmma16(a0, b0, acc00);
    acc01 = wmma16(a0, b1, acc01);
    acc10 = wmma16(a1, b0, acc10);
    acc11 = wmma16(a1, b1, acc11);
    __syncthreads();
  }

  const int nloc = lane & 15;
  const int mhi  = (lane >> 4) << 3;
  const v8f accs[2][2] = {{acc00, acc01}, {acc10, acc11}};
#pragma unroll
  for (int i = 0; i < 2; ++i) {
#pragma unroll
    for (int j = 0; j < 2; ++j) {
      const int ch = g * CG + j * 16 + nloc;
      const float bv = db[ch];
#pragma unroll
      for (int r = 0; r < 8; ++r) {
        const int m = M0 + wv * 32 + i * 16 + mhi + r;
        const int b = m / HWs;
        const int rem = m - b * HWs;
        out[((size_t)b * CCH + ch) * HWs + rem] = accs[i][j][r] + bv;
      }
    }
  }
}

extern "C" void kernel_launch(void* const* d_in, const int* in_sizes, int n_in,
                              void* d_out, int out_size, void* d_ws, size_t ws_size,
                              hipStream_t stream) {
  (void)in_sizes; (void)n_in; (void)out_size; (void)ws_size;
  const float* rgb = (const float*)d_in[0];
  const float* thm = (const float*)d_in[1];
  const float* w1  = (const float*)d_in[2];
  const float* b1  = (const float*)d_in[3];
  const float* w2  = (const float*)d_in[4];
  const float* b2  = (const float*)d_in[5];
  const float* dw  = (const float*)d_in[6];
  const float* db  = (const float*)d_in[7];
  const float* osc = (const float*)d_in[8];
  float* out = (float*)d_out;

  float* hdn  = (float*)d_ws;                                 // [8,64,80,80] f32
  float* offs = hdn + (size_t)BATCH * MID * HWs;              // [8,72,80,80] f32
  _Float16* w1p = (_Float16*)(offs + (size_t)BATCH * OFFC * HWs);
  _Float16* w2p = w1p + 9 * 8 * 64 * 32;
  _Float16* dwp = w2p + 9 * 2 * 2 * 64 * 32;

  dim3 blk(256);
  pack_weights<<<dim3((9 * 8 * 64 * 32 + 255) / 256), blk, 0, stream>>>(
      w1, w2, dw, w1p, w2p, dwp);
  conv3x3_wmma<<<dim3(MTOT / 128, 1), blk, 0, stream>>>(
      rgb, thm, CCH, w1p, b1, nullptr, hdn, 2 * CCH, MID, 1, 1);
  conv3x3_wmma<<<dim3(MTOT / 128, 2), blk, 0, stream>>>(
      hdn, nullptr, MID, w2p, b2, osc, offs, MID, OFFC, 2, 0);
  deform_wmma<<<dim3(MTOT / 256, NG), blk, 0, stream>>>(rgb, offs, dwp, db, out);
}
